// MotherCubeConv_81432579932648
// MI455X (gfx1250) — matrix operand order repaired and checked by reading, not compiled, not run on hardware
//
#include <hip/hip_runtime.h>
#include <hip/hip_bf16.h>

// MotherCubeConv: y = concat(x[self], x[n0..n3]) @ W^T + b
// N=200000, F_in=F_out=128, fan_in=640.
// 5 accumulated bf16 WMMA GEMMs of [128rows,128k]x[128k,128out] per block tile.
// Preferred path: pre-convert x and W to bf16 in d_ws; hot loop is software-
// pipelined async global->LDS gathers (ASYNCcnt, in-order completion) +
// ds_load fragments + v_wmma_f32_16x16x32_bf16.

typedef __attribute__((ext_vector_type(16))) __bf16 v16bf;
typedef __attribute__((ext_vector_type(8)))  float  v8f;
typedef __attribute__((ext_vector_type(4)))  int    v4i;
typedef __attribute__((address_space(1)))    v4i    gv4i;  // global int4
typedef __attribute__((address_space(3)))    v4i    lv4i;  // LDS int4

#define XS 136        // LDS row stride in ushorts (128 + 8 pad -> conflict-free b128)
#define TILE (128 * XS)

#if defined(__HIP_DEVICE_COMPILE__) && __has_builtin(__builtin_amdgcn_global_load_async_to_lds_b128)
#define HAVE_ASYNC_LDS 1
#else
#define HAVE_ASYNC_LDS 0
#endif

__device__ __forceinline__ unsigned short f2bf(float f) {
  unsigned u = __builtin_bit_cast(unsigned, f);
  u += 0x7fffu + ((u >> 16) & 1u);          // round-to-nearest-even
  return (unsigned short)(u >> 16);
}

// 16-byte global->LDS copy; async (ASYNCcnt-tracked) when the builtin exists.
__device__ __forceinline__ void copy16_g2l(const unsigned short* g, unsigned short* l) {
#if HAVE_ASYNC_LDS
  __builtin_amdgcn_global_load_async_to_lds_b128(
      (gv4i*)(uintptr_t)g,
      (lv4i*)(unsigned)(uintptr_t)l, 0, 0);
#else
  *(uint4*)l = *(const uint4*)g;
#endif
}

template <int CNT>
__device__ __forceinline__ void wait_g2l() {
#if HAVE_ASYNC_LDS
#if __has_builtin(__builtin_amdgcn_s_wait_asynccnt)
  __builtin_amdgcn_s_wait_asynccnt(CNT);
#else
  asm volatile("s_wait_asynccnt %0" :: "i"(CNT) : "memory");
#endif
#endif
}

union Frag {
  uint4 u[2];
  v16bf v;
};

// ---- one-shot f32 -> bf16 conversion (x and W into workspace) ----
__global__ void cvt_to_bf16(const float* __restrict__ src,
                            unsigned short* __restrict__ dst, long n4) {
  long i = (long)blockIdx.x * blockDim.x + threadIdx.x;
  if (i < n4) {
    float4 v = ((const float4*)src)[i];
    unsigned lo = (unsigned)f2bf(v.x) | ((unsigned)f2bf(v.y) << 16);
    unsigned hi = (unsigned)f2bf(v.z) | ((unsigned)f2bf(v.w) << 16);
    ((uint2*)dst)[i] = make_uint2(lo, hi);
  }
}

template <bool PRE>
__global__ __launch_bounds__(256, 2)
void mc_conv_wmma(const float* __restrict__ x,
                  const unsigned short* __restrict__ xbf,
                  const int* __restrict__ nbr,
                  const float* __restrict__ W,
                  const unsigned short* __restrict__ wbf,
                  const float* __restrict__ b,
                  float* __restrict__ out,
                  int N) {
  __shared__ __align__(16) unsigned short xT[(PRE ? 2 : 1) * TILE]; // 2x34.8 KB
  __shared__ __align__(16) unsigned short wT[TILE];                 // 34.8 KB

  const int tid   = threadIdx.x;
  const int lane  = tid & 31;
  const int wave  = tid >> 5;
  const int l15   = lane & 15;
  const int lhalf = (lane >> 4) & 1;
  const int rowBase = blockIdx.x * 128;

  v8f acc[8] = {};                           // 8 x (16x16) f32 tiles per wave

  // per-wave async gather: 2 bf16 rows / instruction (16 lanes x 16B)
  auto issue_x = [&](int s, int buf) {
#pragma unroll
    for (int i = 0; i < 8; ++i) {
      const int r = wave * 16 + i * 2 + lhalf;
      int m = rowBase + r;
      if (m >= N) m = N - 1;
      const int src = (s == 0) ? m : nbr[m * 4 + (s - 1)];
      copy16_g2l(xbf + (long)src * 128 + l15 * 8,
                 &xT[buf * TILE + r * XS + l15 * 8]);
    }
  };
  auto issue_w = [&](int s) {
#pragma unroll
    for (int i = 0; i < 8; ++i) {
      const int n = wave * 16 + i * 2 + lhalf;
      copy16_g2l(wbf + (long)n * 640 + s * 128 + l15 * 8,
                 &wT[n * XS + l15 * 8]);
    }
  };

  if constexpr (PRE) issue_x(0, 0);          // pipeline prologue

#pragma unroll
  for (int s = 0; s < 5; ++s) {
    if constexpr (PRE) {
      issue_w(s);                            // wT free: barrier at end of s-1
      if (s < 4) {
        issue_x(s + 1, (s + 1) & 1);         // overlap next gather with compute(s)
        wait_g2l<8>();                       // in-order: x(s), w(s) done; x(s+1) in flight
      } else {
        wait_g2l<0>();
      }
    } else {
      // ---- fallback: f32 loads with inline bf16 conversion ----
      const int c4 = (tid & 31) * 4;
#pragma unroll 4
      for (int r0 = 0; r0 < 128; r0 += 8) {
        const int r = r0 + (tid >> 5);
        int m = rowBase + r;
        if (m >= N) m = N - 1;
        const int src = (s == 0) ? m : nbr[m * 4 + (s - 1)];
        const float4 v = *(const float4*)(x + (long)src * 128 + c4);
        const unsigned lo = (unsigned)f2bf(v.x) | ((unsigned)f2bf(v.y) << 16);
        const unsigned hi = (unsigned)f2bf(v.z) | ((unsigned)f2bf(v.w) << 16);
        *(uint2*)&xT[r * XS + c4] = make_uint2(lo, hi);
      }
#pragma unroll 4
      for (int n0 = 0; n0 < 128; n0 += 8) {
        const int n = n0 + (tid >> 5);
        const float4 v = *(const float4*)(W + (long)n * 640 + s * 128 + c4);
        const unsigned lo = (unsigned)f2bf(v.x) | ((unsigned)f2bf(v.y) << 16);
        const unsigned hi = (unsigned)f2bf(v.z) | ((unsigned)f2bf(v.w) << 16);
        *(uint2*)&wT[n * XS + c4] = make_uint2(lo, hi);
      }
    }
    __syncthreads();

    // ---- WMMA: 16-row strip per wave x 128 cols, K = 128 (4 chunks of 32) ----
    const unsigned short* xb = &xT[PRE ? (s & 1) * TILE : 0];
#pragma unroll
    for (int kc = 0; kc < 4; ++kc) {
      const int kA = kc * 32;
      Frag a;                                // A 16x32 bf16 per ISA 7.12.2
      const int arow = wave * 16 + l15;
      a.u[0] = *(const uint4*)&xb[arow * XS + kA + lhalf * 8];
      a.u[1] = *(const uint4*)&xb[arow * XS + kA + 16 + lhalf * 8];
#pragma unroll
      for (int t = 0; t < 8; ++t) {
        Frag bf;                             // B 32x16: col=l15, K order matches A (8h+s, 16+8h+s)
        const int ncol = t * 16 + l15;
        const int kB = kc * 32 + lhalf * 8;
        bf.u[0] = *(const uint4*)&wT[ncol * XS + kB];
        bf.u[1] = *(const uint4*)&wT[ncol * XS + kB + 16];
        acc[t] = __builtin_amdgcn_wmma_f32_16x16x32_bf16(
            false, a.v, false, bf.v, (short)0, acc[t], false, false);
      }
    }
    __syncthreads();                         // readers done before buffers rewritten
  }

  // ---- bias + store (C layout: VGPR r -> M = r + 8*lhalf, N = l15) ----
  const int rowW = rowBase + wave * 16 + lhalf * 8;
#pragma unroll
  for (int t = 0; t < 8; ++t) {
    const int col = t * 16 + l15;
    const float bias = b[col];
#pragma unroll
    for (int r = 0; r < 8; ++r) {
      const int gm = rowW + r;
      if (gm < N) out[(long)gm * 128 + col] = acc[t][r] + bias;
    }
  }
}

extern "C" void kernel_launch(void* const* d_in, const int* in_sizes, int n_in,
                              void* d_out, int out_size, void* d_ws, size_t ws_size,
                              hipStream_t stream) {
  const float* x   = (const float*)d_in[0];
  const int*   nbr = (const int*)  d_in[1];
  const float* W   = (const float*)d_in[2];
  const float* b   = (const float*)d_in[3];
  float*       out = (float*)d_out;

  const int N = in_sizes[0] / 128;                 // 200000
  const int blocks = (N + 127) / 128;              // 1563 row tiles

  const size_t wBytes = (size_t)128 * 640 * 2;               // 163840
  const size_t wPad   = (wBytes + 255) & ~(size_t)255;
  const size_t xBytes = (size_t)N * 128 * 2;                 // ~51.2 MB

  if (ws_size >= wPad + xBytes) {
    unsigned short* wbf = (unsigned short*)d_ws;
    unsigned short* xbf = (unsigned short*)((char*)d_ws + wPad);
    const long w4 = (long)128 * 640 / 4;
    const long x4 = (long)N * 128 / 4;
    cvt_to_bf16<<<(int)((w4 + 255) / 256), 256, 0, stream>>>(W, wbf, w4);
    cvt_to_bf16<<<(int)((x4 + 255) / 256), 256, 0, stream>>>(x, xbf, x4);
    mc_conv_wmma<true><<<blocks, 256, 0, stream>>>(x, xbf, nbr, W, wbf, b, out, N);
  } else {
    mc_conv_wmma<false><<<blocks, 256, 0, stream>>>(x, nullptr, nbr, W, nullptr, b, out, N);
  }
}
